// LearnableShapelet1D_45724221833337
// MI455X (gfx1250) — matrix operand order repaired and checked by reading, not compiled, not run on hardware
//
#include <hip/hip_runtime.h>
#include <stdint.h>

// Problem constants (from reference): x (16,8,4096) f32, shapelets (64,128) f32,
// output (16,64) f32. T' = 4096-128+1 = 3969.
#define B_   16
#define C_   8
#define T_   4096
#define K_   64
#define L_   128
#define TP_  3969

typedef __bf16 bf16_t;
typedef __attribute__((ext_vector_type(16))) bf16_t bf16x16;
typedef __attribute__((ext_vector_type(8)))  float  v8f;

union ABf {                 // one WMMA 16x32 bf16 operand, per-lane slice (8 VGPRs)
  bf16x16        v;
  unsigned short s[16];
  uint32_t       u[8];
  uint4          q[2];
};

__device__ __forceinline__ unsigned short f32_to_bf16_rte(float f) {
  uint32_t u = __float_as_uint(f);
  uint32_t r = u + 0x7FFFu + ((u >> 16) & 1u);   // round-to-nearest-even
  return (unsigned short)(r >> 16);
}
__device__ __forceinline__ float bf16us_to_f32(unsigned short h) {
  return __uint_as_float(((uint32_t)h) << 16);
}
// IEEE-f32 -> monotonic unsigned key (total order), so atomicMin(u32) == float min.
__device__ __forceinline__ uint32_t enc_key(float f) {
  uint32_t u = __float_as_uint(f);
  return u ^ ((u & 0x80000000u) ? 0xFFFFFFFFu : 0x80000000u);
}
__device__ __forceinline__ float dec_key(uint32_t k) {
  uint32_t u = (k & 0x80000000u) ? (k ^ 0x80000000u) : ~k;
  return __uint_as_float(u);
}

// ---- Kernel 1: fold channels AND pre-split to bf16 hi/lo.
// xs = sum_c x  -> stored only as (hi,lo) bf16 pair; xq = sum_c x^2 (exact f32).
__global__ __launch_bounds__(256) void prep_x(const float* __restrict__ x,
                                              unsigned short* __restrict__ xs_hi,
                                              unsigned short* __restrict__ xs_lo,
                                              float* __restrict__ xq) {
  int idx = blockIdx.x * 256 + threadIdx.x;          // 16*4096 = 65536 threads
  if (idx >= B_ * T_) return;
  int b = idx >> 12, t = idx & (T_ - 1);
  const float* p = x + (size_t)b * C_ * T_ + t;
  float s = 0.f, q = 0.f;
#pragma unroll
  for (int c = 0; c < C_; ++c) { float v = p[(size_t)c * T_]; s += v; q += v * v; }
  unsigned short h = f32_to_bf16_rte(s);
  xs_hi[idx] = h;
  xs_lo[idx] = f32_to_bf16_rte(s - bf16us_to_f32(h));
  xq[idx] = q;
}

// ---- Kernel 2: shapelet hi/lo bf16 split + exact ||s_k||^2 ----
__global__ __launch_bounds__(128) void prep_s(const float* __restrict__ sh,
                                              unsigned short* __restrict__ s_hi,
                                              unsigned short* __restrict__ s_lo,
                                              float* __restrict__ s2) {
  __shared__ float red[128];
  int k = blockIdx.x, l = threadIdx.x;
  float v = sh[k * L_ + l];
  unsigned short h = f32_to_bf16_rte(v);
  s_hi[k * L_ + l] = h;
  s_lo[k * L_ + l] = f32_to_bf16_rte(v - bf16us_to_f32(h));
  red[l] = v * v;
  __syncthreads();
  for (int s = 64; s > 0; s >>= 1) { if (l < s) red[l] += red[l + s]; __syncthreads(); }
  if (l == 0) s2[k] = red[0];
}

__global__ void init_keys(uint32_t* keys) {
  int i = blockIdx.x * 256 + threadIdx.x;
  if (i < B_ * K_) keys[i] = 0xFFFFFFFFu;
}
__global__ void decode_keys(const uint32_t* __restrict__ keys, float* __restrict__ out) {
  int i = blockIdx.x * 256 + threadIdx.x;
  if (i < B_ * K_) out[i] = dec_key(keys[i]);
}

// ---- Kernel 3: main WMMA kernel.
// Block = 256 thr (8 waves): batch b, 32 time positions t0..t0+31, all 64 shapelets.
// Wave w: kt = w&3 (16-shapelet tile), mloc = (w>>2)*16 (which 16-t half).
// A operands gathered as pre-packed bf16 pairs from two parity-staggered LDS copies
// (copy1 shifted by one element), so every lane issues aligned dword DS loads.
__global__ __launch_bounds__(256) void shapelet_main(
    const unsigned short* __restrict__ xs_hi, const unsigned short* __restrict__ xs_lo,
    const float* __restrict__ xq,
    const unsigned short* __restrict__ s_hi, const unsigned short* __restrict__ s_lo,
    const float* __restrict__ s2, uint32_t* __restrict__ keys) {
  __shared__ uint32_t hi0_u[80], hi1_u[80], lo0_u[80], lo1_u[80];  // 160 u16 each
  __shared__ float xq_sh[160];
  __shared__ float w_part[256];
  __shared__ float w_sh[32];

  const int b    = blockIdx.y;
  const int t0   = blockIdx.x * 32;
  const int tid  = threadIdx.x;
  const int wave = tid >> 5;
  const int lane = tid & 31;
  const int half = lane >> 4;
  const int n    = lane & 15;          // N index (shapelet within tile) and A row m
  const int kt   = wave & 3;
  const int mloc = (wave >> 2) * 16;   // 0 or 16 within the block's 32-t tile

  // Resident B tiles: 4 K-chunks x (hi,lo). Lane holds s[kt*16+n, 32c+8*half .. +7] and [32c+16+8*half .. +7].
  ABf bh[4], bl[4];
  {
    const int row = kt * 16 + n;
#pragma unroll
    for (int c = 0; c < 4; ++c) {
      int off = row * L_ + c * 32 + half * 8;            // 16B-aligned (8 shorts)
      bh[c].q[0] = *(const uint4*)(s_hi + off);
      bh[c].q[1] = *(const uint4*)(s_hi + off + 16);
      bl[c].q[0] = *(const uint4*)(s_lo + off);
      bl[c].q[1] = *(const uint4*)(s_lo + off + 16);
    }
  }

  // Stage xs hi/lo (two staggered copies) + xq. Tail guarded (t0+158 may exceed 4095).
  {
    unsigned short* hi0 = (unsigned short*)hi0_u;
    unsigned short* hi1 = (unsigned short*)hi1_u;
    unsigned short* lo0 = (unsigned short*)lo0_u;
    unsigned short* lo1 = (unsigned short*)lo1_u;
    if (tid < 160) {
      int idx = t0 + tid;
      unsigned short vh = 0, vl = 0; float vq = 0.f;
      if (idx < T_) {
        vh = xs_hi[b * T_ + idx];
        vl = xs_lo[b * T_ + idx];
        vq = xq[b * T_ + idx];
      }
      hi0[tid] = vh; lo0[tid] = vl; xq_sh[tid] = vq;
      if (tid > 0) { hi1[tid - 1] = vh; lo1[tid - 1] = vl; }  // copy shifted by 1 elem
    }
  }
  __syncthreads();

  // Sliding ||window||^2: W(t) = sum_{l<128} xq[t+l], parallel over all 256 threads
  // (each sums 16, then an 8-way fold).  t = tid>>3, seg = tid&7.
  {
    float p = 0.f;
    int base = (tid >> 3) + (tid & 7) * 16;
#pragma unroll
    for (int j = 0; j < 16; ++j) p += xq_sh[base + j];
    w_part[tid] = p;
  }
  __syncthreads();
  if (tid < 32) {
    float a = 0.f;
#pragma unroll
    for (int j = 0; j < 8; ++j) a += w_part[tid * 8 + j];
    w_sh[tid] = a;
  }
  __syncthreads();

  // GEMM: acc(16t x 16k) += A(16x32)*B(32x16), 4 K-chunks x 3 split terms.
  // ISA 16-bit A layout: lanes<16 hold K=0..7 (v0..3) & 16..23 (v4..7);
  // lanes>=16 hold K=8..15 & 24..31; row m = lane&15. Element j of chunk c for this
  // lane is xs[t0 + mloc + n + 32c + 8*half + j] -> aligned dwords in parity copy.
  const int par = n & 1;
  const uint32_t* hp = par ? hi1_u : hi0_u;
  const uint32_t* lp = par ? lo1_u : lo0_u;
  v8f acc = {};
#pragma unroll
  for (int c = 0; c < 4; ++c) {
    ABf ah, al;
    int e0 = mloc + n + c * 32 + half * 8;     // first element (parity == par)
    int dw = (e0 - par) >> 1;                  // aligned dword index in parity copy
#pragma unroll
    for (int r = 0; r < 4; ++r) {
      ah.u[r]     = hp[dw + r];                // K = 2r, 2r+1
      ah.u[4 + r] = hp[dw + 8 + r];            // K = 16+2r, 16+2r+1
      al.u[r]     = lp[dw + r];
      al.u[4 + r] = lp[dw + 8 + r];
    }
    acc = __builtin_amdgcn_wmma_f32_16x16x32_bf16(false, ah.v, false, bh[c].v,
                                                  (short)0, acc, false, false);
    acc = __builtin_amdgcn_wmma_f32_16x16x32_bf16(false, ah.v, false, bl[c].v,
                                                  (short)0, acc, false, false);
    acc = __builtin_amdgcn_wmma_f32_16x16x32_bf16(false, al.v, false, bh[c].v,
                                                  (short)0, acc, false, false);
  }

  // Epilogue: dist = W(t) - 2*corr + C*||s||^2 ; min over t via monotonic-key atomicMin.
  const float s2v = s2[kt * 16 + n] * (float)C_;
  uint32_t kmin = 0xFFFFFFFFu;
#pragma unroll
  for (int r = 0; r < 8; ++r) {
    int mo = r + 8 * half;             // D layout: VGPR r -> M=r (lanes<16), M=8+r (lanes>=16)
    int t  = t0 + mloc + mo;
    float dist = w_sh[mloc + mo] - 2.0f * acc[r] + s2v;
    uint32_t key = (t < TP_) ? enc_key(dist) : 0xFFFFFFFFu;
    kmin = key < kmin ? key : kmin;
  }
  uint32_t other = __shfl_xor(kmin, 16, 32);   // fold the two M-halves (same n -> same k)
  kmin = other < kmin ? other : kmin;
  if (half == 0) atomicMin(&keys[b * K_ + kt * 16 + n], kmin);
}

extern "C" void kernel_launch(void* const* d_in, const int* in_sizes, int n_in,
                              void* d_out, int out_size, void* d_ws, size_t ws_size,
                              hipStream_t stream) {
  (void)in_sizes; (void)n_in; (void)out_size; (void)ws_size;
  const float* x  = (const float*)d_in[0];   // (16,8,4096) f32
  const float* sh = (const float*)d_in[1];   // (64,128)    f32
  float* out = (float*)d_out;                // (16,64)     f32

  // Workspace carve-up (~560 KB total).
  float*          xq   = (float*)d_ws;                        // 65536 f32
  unsigned short* xshi = (unsigned short*)(xq + B_ * T_);     // 65536 u16
  unsigned short* xslo = xshi + B_ * T_;                      // 65536 u16
  unsigned short* s_hi = xslo + B_ * T_;                      // 8192 u16
  unsigned short* s_lo = s_hi + K_ * L_;                      // 8192 u16
  float*          s2   = (float*)(s_lo + K_ * L_);            // 64 f32
  uint32_t*       keys = (uint32_t*)(s2 + K_);                // 1024 u32

  prep_x   <<<dim3((B_ * T_ + 255) / 256), 256, 0, stream>>>(x, xshi, xslo, xq);
  prep_s   <<<dim3(K_), 128, 0, stream>>>(sh, s_hi, s_lo, s2);
  init_keys<<<dim3(4), 256, 0, stream>>>(keys);
  shapelet_main<<<dim3((TP_ + 31) / 32, B_), 256, 0, stream>>>(xshi, xslo, xq,
                                                               s_hi, s_lo, s2, keys);
  decode_keys<<<dim3(4), 256, 0, stream>>>(keys, out);
}
